// SelfAttention_57380763074621
// MI455X (gfx1250) — matrix operand order repaired and checked by reading, not compiled, not run on hardware
//
#include <hip/hip_runtime.h>
#include <math.h>

// ---------------------------------------------------------------------------
// Types for CDNA5 WMMA (wave32): D(16x16 f32) = A(16x32 bf16) x B(32x16 bf16) + C
// ---------------------------------------------------------------------------
typedef __attribute__((ext_vector_type(16))) __bf16 v16bf;
typedef __attribute__((ext_vector_type(8)))  __bf16 v8bf;
typedef __attribute__((ext_vector_type(8)))  float  v8f;
typedef __attribute__((ext_vector_type(4)))  int    v4i;

// address-space-qualified int4 pointee types for the async-copy builtin
typedef __attribute__((address_space(1))) v4i gv4i;  // global
typedef __attribute__((address_space(3))) v4i sv4i;  // LDS

#define WMMA_BF16(a, b, c) \
    __builtin_amdgcn_wmma_f32_16x16x32_bf16(false, (a), false, (b), (short)0, (c), false, false)

// Round-to-nearest-even fp32 -> bf16 (raw bits)
__device__ __forceinline__ unsigned short f2bf_raw(float f) {
    union { float f; unsigned u; } v; v.f = f;
    unsigned r = v.u + 0x7FFFu + ((v.u >> 16) & 1u);
    return (unsigned short)(r >> 16);
}

// ---------------------------------------------------------------------------
// CDNA5 async global->LDS copy (ASYNCcnt-tracked, no VGPR data round trip).
// LDS operand is the low 32 bits of the generic pointer (ISA: LDS aperture
// addresses carry the LDS byte offset in addr[31:0]).
// ---------------------------------------------------------------------------
#if __has_builtin(__builtin_amdgcn_global_load_async_to_lds_b128)
#define ASYNC_VIA_BUILTIN 1
#endif

__device__ __forceinline__ void async_cp_b128(const void* g, void* l) {
#ifdef ASYNC_VIA_BUILTIN
    __builtin_amdgcn_global_load_async_to_lds_b128(
        (gv4i*)(unsigned long long)g,
        (sv4i*)(unsigned)(unsigned long long)l,
        0, 0);
#else
    asm volatile("global_load_async_to_lds_b128 %0, %1, off"
                 :: "v"((unsigned)(unsigned long long)l), "v"(g)
                 : "memory");
#endif
}

__device__ __forceinline__ void wait_async0() {
#if __has_builtin(__builtin_amdgcn_s_wait_asynccnt)
    __builtin_amdgcn_s_wait_asynccnt(0);
#else
    asm volatile("s_wait_asynccnt 0x0" ::: "memory");
#endif
}

// A-fragment (16x32, 16-bit): lane m = lane%16, half = lane/16.
// VGPRs 0-3 hold K = half*8 + 0..7 ; VGPRs 4-7 hold K = 16 + half*8 + 0..7.
__device__ __forceinline__ v16bf load_a16(const unsigned short* lo) {
    union { v16bf v; v8bf h[2]; } u;
    u.h[0] = *(const v8bf*)lo;
    u.h[1] = *(const v8bf*)(lo + 16);
    return u.v;
}

// B-fragment (32x16, 16-bit): lane n = lane%16, half = lane/16.
// VGPRs 0-3 hold K = half*8 + 0..7 ; VGPRs 4-7 hold K = 16 + half*8 + 0..7.
__device__ __forceinline__ v16bf load_b16(const unsigned short* p) {
    const unsigned short* lo = p - ((threadIdx.x & 31) >> 4) * 8;
    union { v16bf v; v8bf h[2]; } u;
    u.h[0] = *(const v8bf*)lo;
    u.h[1] = *(const v8bf*)(lo + 16);
    return u.v;
}

// ---------------------------------------------------------------------------
// Kernel 0: fp32 -> bf16 elementwise (weights)
// ---------------------------------------------------------------------------
__global__ void wconv_kernel(const float* __restrict__ src,
                             unsigned short* __restrict__ dst, int n) {
    int i = (blockIdx.x * blockDim.x + threadIdx.x) * 8;
    if (i >= n) return;
    float4 f0 = *(const float4*)(src + i);
    float4 f1 = *(const float4*)(src + i + 4);
    uint4 o;
    o.x = (unsigned)f2bf_raw(f0.x) | ((unsigned)f2bf_raw(f0.y) << 16);
    o.y = (unsigned)f2bf_raw(f0.z) | ((unsigned)f2bf_raw(f0.w) << 16);
    o.z = (unsigned)f2bf_raw(f1.x) | ((unsigned)f2bf_raw(f1.y) << 16);
    o.w = (unsigned)f2bf_raw(f1.z) | ((unsigned)f2bf_raw(f1.w) << 16);
    *(uint4*)(dst + i) = o;
}

// ---------------------------------------------------------------------------
// Kernel 1: q/k/v = x @ W^T + b (bf16 out). v is written TRANSPOSED [512][8192]
// so the attention kernel can stage it with pure async b128 copies.
// Block: 64 x-rows, 512 threads = 16 waves; wave (rt=wid&3, ch=wid>>2) owns
// rows [rt*16,+16) x cols [ch*128,+128). W chunks (512x32) double-buffered in
// LDS via async copies. LDS: x 64KB + 2*32KB = 128KB.
// ---------------------------------------------------------------------------
__global__ __launch_bounds__(512) void qkv_kernel(
    const float* __restrict__ x,
    const unsigned short* __restrict__ Wq, const unsigned short* __restrict__ Wk,
    const unsigned short* __restrict__ Wv,
    const float* __restrict__ bq, const float* __restrict__ bk,
    const float* __restrict__ bv,
    unsigned short* __restrict__ qo, unsigned short* __restrict__ ko,
    unsigned short* __restrict__ vo /* transposed [512][8192] */) {
    extern __shared__ char smem[];
    unsigned short* lds_x = (unsigned short*)smem;   // [64][512]
    unsigned short* lds_w0 = lds_x + 64 * 512;       // [512][32] buf 0
    unsigned short* lds_w1 = lds_w0 + 512 * 32;      // [512][32] buf 1

    const int tid = threadIdx.x;
    const int wid = tid >> 5, lane = tid & 31;
    const int half = lane >> 4, ln = lane & 15;
    const int rt = wid & 3, ch = wid >> 2;
    const long xbase = (long)blockIdx.x * (64 * 512);

    const unsigned short* Ws[3] = {Wq, Wk, Wv};
    const float* Bs[3] = {bq, bk, bv};
    unsigned short* Os[3] = {qo, ko, vo};

    // stage one W K-chunk (cols kk*32..+32 of rows 0..255) via async copies
    auto stage_w = [&](const unsigned short* W, int kk, unsigned short* buf) {
#pragma unroll
        for (int i = 0; i < 2; ++i) {
            int id = i * 512 + tid;              // [0,1024) uint4 chunks
            int o = id >> 2, off = (id & 3) * 8;
            async_cp_b128(W + (long)o * 512 + kk * 32 + off, buf + o * 32 + off);
        }
    };

    // cooperative load x (fp32) -> bf16 in LDS
    for (int i = 0; i < 16; ++i) {
        int e = (i * 512 + tid) * 4;
        float4 f = *(const float4*)(x + xbase + e);
        unsigned p0 = (unsigned)f2bf_raw(f.x) | ((unsigned)f2bf_raw(f.y) << 16);
        unsigned p1 = (unsigned)f2bf_raw(f.z) | ((unsigned)f2bf_raw(f.w) << 16);
        *(uint2*)(lds_x + e) = make_uint2(p0, p1);
    }

    stage_w(Ws[0], 0, lds_w0);  // preload chunk 0

    v8f c[8];
    for (int cc = 0; cc < 48; ++cc) {  // 3 mats x 16 K-chunks, linearized
        const int mat = cc >> 4, kk = cc & 15;
        if (kk == 0)
            for (int i = 0; i < 8; ++i) c[i] = {};

        wait_async0();       // chunk cc landed (this wave's share)
        __syncthreads();     // all waves' shares landed; prev buffer dead
        if (cc + 1 < 48)     // prefetch next chunk into other buffer
            stage_w(Ws[(cc + 1) >> 4], (cc + 1) & 15,
                    ((cc + 1) & 1) ? lds_w1 : lds_w0);

        unsigned short* wbuf = (cc & 1) ? lds_w1 : lds_w0;
        v16bf a = load_a16(lds_x + (rt * 16 + ln) * 512 + kk * 32 + half * 8);
#pragma unroll
        for (int ct = 0; ct < 8; ++ct) {
            int o = ch * 128 + ct * 16 + ln;
            v16bf b = load_b16(wbuf + o * 32 + half * 16);  // B = W^T
            c[ct] = WMMA_BF16(a, b, c[ct]);
        }

        if (kk == 15) {  // epilogue: + bias, -> bf16, store (v transposed)
            const float* bias = Bs[mat];
            unsigned short* og = Os[mat];
#pragma unroll
            for (int ct = 0; ct < 8; ++ct) {
                int o = ch * 128 + ct * 16 + ln;
                float bv2 = bias[o];
#pragma unroll
                for (int r = 0; r < 8; ++r) {
                    long row = (long)blockIdx.x * 64 + rt * 16 + half * 8 + r;
                    unsigned short val = f2bf_raw(c[ct][r] + bv2);
                    if (mat == 2) og[(long)o * 8192 + row] = val;
                    else          og[row * 512 + o] = val;
                }
            }
        }
    }
}

// ---------------------------------------------------------------------------
// Kernel 2: out = (q k^T) v / max(||row||_2, eps), streamed over j.
// Block: 64 q-rows, 512 threads = 16 waves; wave (rt, ch) owns out slab
// rows [rt*16,+16) x cols [ch*128,+128). Per 64-row j-tile:
//   stage1: 16 WMMAs -> one 16x16 s tile / wave; s^2 into regs; s -> LDS bf16.
//   stage2: 2x8 WMMAs accumulate s @ v into out slab.
// All staging is async global->LDS: v^T tile double-buffered (prefetched at
// tile top, overlapping both stages); next k tile issued right after stage 1
// (lds_k dead). LDS: q 64 + k 64 + 2*v^T 64 + s 8 KB + norms = 264.25KB.
// ---------------------------------------------------------------------------
__global__ __launch_bounds__(512) void attn_kernel(
    const unsigned short* __restrict__ qg, const unsigned short* __restrict__ kg,
    const unsigned short* __restrict__ vtg /* [512][8192] */,
    float* __restrict__ out) {
    extern __shared__ char smem[];
    unsigned short* lds_q   = (unsigned short*)smem;  // [64][512]
    unsigned short* lds_k   = lds_q + 64 * 512;       // [64][512]
    unsigned short* lds_vt0 = lds_k + 64 * 512;       // [512][64]  buf 0
    unsigned short* lds_vt1 = lds_vt0 + 512 * 64;     // [512][64]  buf 1
    unsigned short* lds_s   = lds_vt1 + 512 * 64;     // [4][16][64]
    float* lds_nrm = (float*)(lds_s + 4 * 16 * 64);   // [64]

    const int tid = threadIdx.x;
    const int wid = tid >> 5, lane = tid & 31;
    const int half = lane >> 4, ln = lane & 15;
    const int rt = wid & 3, ch = wid >> 2;
    const long qbase = (long)blockIdx.x * (64 * 512);

    auto stage_k = [&](int jt) {  // k rows jt*64..+64, row-major
#pragma unroll
        for (int i = 0; i < 8; ++i) {
            int idx = (i * 512 + tid) * 8;
            async_cp_b128(kg + (long)jt * (64 * 512) + idx, lds_k + idx);
        }
    };
    auto stage_vt = [&](int jt, unsigned short* buf) {  // v^T cols jt*64..+64
#pragma unroll
        for (int i = 0; i < 8; ++i) {
            int id = i * 512 + tid;               // [0,4096) uint4 chunks
            int row = id >> 3, off = (id & 7) * 8;
            async_cp_b128(vtg + (long)row * 8192 + (long)jt * 64 + off,
                          buf + row * 64 + off);
        }
    };

    if (tid < 64) lds_nrm[tid] = 0.0f;
#pragma unroll
    for (int i = 0; i < 8; ++i) {  // q tile -> LDS (async)
        int idx = (i * 512 + tid) * 8;
        async_cp_b128(qg + qbase + idx, lds_q + idx);
    }
    stage_k(0);
    stage_vt(0, lds_vt0);

    float na[8];
#pragma unroll
    for (int r = 0; r < 8; ++r) na[r] = 0.0f;
    v8f c[8];
    for (int i = 0; i < 8; ++i) c[i] = {};

    for (int jt = 0; jt < 128; ++jt) {
        wait_async0();    // q/k(jt)/vt(jt) (and nothing else) complete
        __syncthreads();  // visible to all waves; old buffers dead
        if (jt + 1 < 128)  // prefetch next v^T tile; overlaps both stages
            stage_vt(jt + 1, ((jt + 1) & 1) ? lds_vt1 : lds_vt0);

        // ---- stage 1: s tile (rows rt*16.., j cols jt*64 + ch*16..), K=512
        v8f s = {};
#pragma unroll
        for (int kk = 0; kk < 16; ++kk) {
            v16bf a = load_a16(lds_q + (rt * 16 + ln) * 512 + kk * 32 + half * 8);
            v16bf b = load_b16(lds_k + (ch * 16 + ln) * 512 + kk * 32 + half * 16);
            s = WMMA_BF16(a, b, s);
        }
#pragma unroll
        for (int r = 0; r < 8; ++r) {  // norm acc + C-layout -> row-major bf16
            float sv = s[r];
            na[r] += sv * sv;
            lds_s[rt * 1024 + (half * 8 + r) * 64 + ch * 16 + ln] = f2bf_raw(sv);
        }
        __syncthreads();   // s complete; lds_k dead
        if (jt + 1 < 128)  // prefetch next k tile; overlaps stage 2
            stage_k(jt + 1);

        // ---- stage 2: out slab += s(16x64) @ v(64x128)
        unsigned short* vt_cur = (jt & 1) ? lds_vt1 : lds_vt0;
#pragma unroll
        for (int kc = 0; kc < 2; ++kc) {
            v16bf a2 = load_a16(lds_s + rt * 1024 + ln * 64 + kc * 32 + half * 8);
#pragma unroll
            for (int ct = 0; ct < 8; ++ct) {
                int col = ch * 128 + ct * 16 + ln;
                v16bf b2 = load_b16(vt_cur + col * 64 + kc * 32 + half * 16);
                c[ct] = WMMA_BF16(a2, b2, c[ct]);
            }
        }
    }

    // ---- reduce row norms, scale, store fp32
    __syncthreads();
#pragma unroll
    for (int r = 0; r < 8; ++r)
        atomicAdd(&lds_nrm[rt * 16 + half * 8 + r], na[r]);
    __syncthreads();
    float inv[8];
#pragma unroll
    for (int r = 0; r < 8; ++r) {
        float nrm = sqrtf(lds_nrm[rt * 16 + half * 8 + r]);
        inv[r] = 1.0f / fmaxf(nrm, 1e-12f);
    }
#pragma unroll
    for (int ct = 0; ct < 8; ++ct) {
        int col = ch * 128 + ct * 16 + ln;
#pragma unroll
        for (int r = 0; r < 8; ++r) {
            long row = (long)blockIdx.x * 64 + rt * 16 + half * 8 + r;
            out[row * 512 + col] = c[ct][r] * inv[r];
        }
    }
}

// ---------------------------------------------------------------------------
// Host launch. Workspace (bf16 as ushort): Wq/Wk/Wv 3*512*512, q/k 8192*512
// each, v^T 512*8192  (~25.6 MB total).
// ---------------------------------------------------------------------------
extern "C" void kernel_launch(void* const* d_in, const int* in_sizes, int n_in,
                              void* d_out, int out_size, void* d_ws, size_t ws_size,
                              hipStream_t stream) {
    (void)in_sizes; (void)n_in; (void)out_size; (void)ws_size;
    const float* x  = (const float*)d_in[0];
    const float* Wq = (const float*)d_in[1];
    const float* bq = (const float*)d_in[2];
    const float* Wk = (const float*)d_in[3];
    const float* bk = (const float*)d_in[4];
    const float* Wv = (const float*)d_in[5];
    const float* bv = (const float*)d_in[6];
    float* out = (float*)d_out;

    unsigned short* Wqb = (unsigned short*)d_ws;
    unsigned short* Wkb = Wqb + 512 * 512;
    unsigned short* Wvb = Wkb + 512 * 512;
    unsigned short* qb  = Wvb + 512 * 512;
    unsigned short* kb  = qb + 8192 * 512;
    unsigned short* vtb = kb + 8192 * 512;  // transposed [512][8192]

    const int qkv_lds  = (64 * 512 + 2 * 512 * 32) * 2;                   // 128 KB
    const int attn_lds = (64 * 512 * 2 + 2 * 512 * 64 + 4096) * 2 + 256;  // 264.25 KB
    (void)hipFuncSetAttribute((const void*)qkv_kernel,
                              hipFuncAttributeMaxDynamicSharedMemorySize, qkv_lds);
    (void)hipFuncSetAttribute((const void*)attn_kernel,
                              hipFuncAttributeMaxDynamicSharedMemorySize, attn_lds);

    wconv_kernel<<<128, 256, 0, stream>>>(Wq, Wqb, 512 * 512);
    wconv_kernel<<<128, 256, 0, stream>>>(Wk, Wkb, 512 * 512);
    wconv_kernel<<<128, 256, 0, stream>>>(Wv, Wvb, 512 * 512);

    qkv_kernel<<<128, 512, qkv_lds, stream>>>(x, Wqb, Wkb, Wvb, bq, bk, bv,
                                              qb, kb, vtb);
    attn_kernel<<<128, 512, attn_lds, stream>>>(qb, kb, vtb, out);
}
